// WrapperTransformerConv1d_10024453669436
// MI455X (gfx1250) — matrix operand order repaired and checked by reading, not compiled, not run on hardware
//
#include <hip/hip_runtime.h>
#include <hip/hip_bf16.h>

// ---------------------------------------------------------------------------
// Problem: x [4096,2048] f32, weight_t [8192,2048] f32, bias [8192],
// groupwise 4-bit fake quant (GS=128 along IN), out = bias + x @ wq.T.
// GEMM (137 GFLOP) dominates -> bf16 WMMA with TDM-fed double-buffered LDS.
// ---------------------------------------------------------------------------
#define IN_F   2048
#define NF     8192
#define MROWS  4096
#define GS     128
#define NGRP   (NF * (IN_F / GS))
#define MAXQ   15.0f

#define KT     64                 // K-tile (2 WMMA K-steps)
#define LDSS   72                 // LDS row stride in bf16 elems (128B data + 16B pad)

typedef __bf16 bf16_t;
typedef __attribute__((ext_vector_type(16))) __bf16 v16bf;
typedef __attribute__((ext_vector_type(8)))  __bf16 v8bf;
typedef __attribute__((ext_vector_type(4)))  __bf16 v4bf;
typedef __attribute__((ext_vector_type(8)))  float  v8f;
typedef __attribute__((ext_vector_type(4)))  float  v4f;
typedef __attribute__((ext_vector_type(4)))  unsigned int v4u;
typedef __attribute__((ext_vector_type(8)))  int    v8i;
typedef __attribute__((ext_vector_type(4)))  int    v4i;

// ---------------------------------------------------------------------------
// Kernel 1: groupwise 4-bit fake-quant of weight_t -> bf16 (1 wave / group).
// ---------------------------------------------------------------------------
__global__ __launch_bounds__(256) void quant_w_kernel(
    const float* __restrict__ w, const float* __restrict__ val,
    const float* __restrict__ mins, const float* __restrict__ maxs,
    bf16_t* __restrict__ wq)
{
    const int lane = threadIdx.x & 31;
    const int wv   = threadIdx.x >> 5;
    const long g   = (long)blockIdx.x * 8 + wv;
    const long base = g * GS + (long)lane * 4;

    v4f wvec = *(const v4f*)(w + base);
    v4f vvec = *(const v4f*)(val + base);

    float mn = fminf(fminf(wvec[0], wvec[1]), fminf(wvec[2], wvec[3]));
    float mx = fmaxf(fmaxf(wvec[0], wvec[1]), fmaxf(wvec[2], wvec[3]));
    #pragma unroll
    for (int m = 16; m >= 1; m >>= 1) {
        mn = fminf(mn, __shfl_xor(mn, m, 32));
        mx = fmaxf(mx, __shfl_xor(mx, m, 32));
    }

    float ms = fminf(fmaxf(mins[g], -1.0f), 0.0f);
    float xs = fminf(fmaxf(maxs[g], -1.0f), 0.0f);

    float wmin = fminf(mn, 0.0f) * (ms + 1.0f);
    float wmax = fmaxf(mx, 0.0f) * (xs + 1.0f);
    if (wmin == 0.0f && wmax == 0.0f) { wmin = -1.0f; wmax = 1.0f; }

    float scale = (wmax - wmin) * (1.0f / MAXQ);
    float zp    = rintf(-wmin / scale);

    v4bf o;
    #pragma unroll
    for (int i = 0; i < 4; ++i) {
        float iw = rintf(wvec[i] / scale + vvec[i]);
        float q  = fminf(fmaxf(iw + zp, 0.0f), MAXQ);
        o[i] = (bf16_t)(scale * (q - zp));
    }
    *(v4bf*)(wq + base) = o;
}

// ---------------------------------------------------------------------------
// Kernel 2: x f32 -> bf16 (8 elems / thread).
// ---------------------------------------------------------------------------
__global__ __launch_bounds__(256) void cvt_x_kernel(
    const float* __restrict__ x, bf16_t* __restrict__ xb)
{
    const long i = ((long)blockIdx.x * 256 + threadIdx.x) * 8;
    v4f a = *(const v4f*)(x + i);
    v4f b = *(const v4f*)(x + i + 4);
    v8bf o;
    #pragma unroll
    for (int j = 0; j < 4; ++j) { o[j] = (bf16_t)a[j]; o[4 + j] = (bf16_t)b[j]; }
    *(v8bf*)(xb + i) = o;
}

// ---------------------------------------------------------------------------
// TDM: issue one 2-D tile load (128 rows x 64 bf16) global -> LDS with
// per-row padding (128B data + 16B pad => LDS row stride 144B = LDSS elems).
// D# layout per CDNA5 ISA.
// This toolchain exposes the 6-arg builtin:
//   (uint32x4 g0, int32x8 g1, int32x4 g2, int32x4 g3, int32x8 g4, i32 cpol)
// ---------------------------------------------------------------------------
__device__ __forceinline__ void tdm_load_tile(unsigned long long gaddr,
                                              unsigned lds_addr,
                                              unsigned tensor_dim1)
{
    v4u g0;
    g0[0] = 1u;                                            // count=1 (valid), user mode
    g0[1] = lds_addr;                                      // LDS byte address
    g0[2] = (unsigned)(gaddr & 0xFFFFFFFFu);               // global_addr[31:0]
    g0[3] = (unsigned)((gaddr >> 32) & 0x01FFFFFFu)        // global_addr[56:32]
          | (2u << 30);                                    // type=2 ("image")

    v8i g1;
    g1[0] = (int)((1u << 16)        // data_size = 1 (2-byte elements)
                | (1u << 20)        // pad_enable
                | (4u << 22)        // pad_interval code 4 = 32 DWORDs (128B)
                | (3u << 25));      // pad_amount  code 3 = 4 DWORDs (16B)
    g1[1] = (int)(((unsigned)IN_F & 0xFFFFu) << 16);       // tensor_dim0[15:0]
    g1[2] = (int)((((unsigned)IN_F >> 16) & 0xFFFFu)       // tensor_dim0[31:16]
                | ((tensor_dim1 & 0xFFFFu) << 16));        // tensor_dim1[15:0]
    g1[3] = (int)(((tensor_dim1 >> 16) & 0xFFFFu)          // tensor_dim1[31:16]
                | ((unsigned)KT << 16));                   // tile_dim0 = 64
    g1[4] = 128;                                           // tile_dim1=128, tile_dim2=0
    g1[5] = IN_F;                                          // tensor_dim0_stride[31:0]
    g1[6] = 0;                                             // stride0 hi / stride1 lo
    g1[7] = 0;

    v4i g2 = {0, 0, 0, 0};
    v4i g3 = {0, 0, 0, 0};
    v8i g4 = {0, 0, 0, 0, 0, 0, 0, 0};
    __builtin_amdgcn_tensor_load_to_lds(g0, g1, g2, g3, g4, 0);
}

// ---------------------------------------------------------------------------
// Kernel 3: bf16 WMMA GEMM, TDM double-buffered.
// Block tile 128x128, 8 waves (2x4), wave tile 64x32 = 4x2 accumulators.
// 32 K-iterations of KT=64 (16 WMMAs / wave / iteration).
// ---------------------------------------------------------------------------
__global__ __launch_bounds__(256) void gemm_bf16_kernel(
    const bf16_t* __restrict__ A,     // [MROWS, IN_F]
    const bf16_t* __restrict__ Bw,    // [NF, IN_F]
    const float*  __restrict__ bias,  // [NF]
    float* __restrict__ out)          // [MROWS, NF]
{
    __shared__ __align__(16) bf16_t lsA[2][128 * LDSS];
    __shared__ __align__(16) bf16_t lsB[2][128 * LDSS];

    const int tid   = threadIdx.x;
    const int lane  = tid & 31;
    const int wave  = tid >> 5;
    const int waveM = wave >> 2;        // 0..1
    const int waveN = wave & 3;         // 0..3
    const int half  = lane >> 4;        // ISA lane-group (0/1)
    const int l16   = lane & 15;

    const long mBlock = (long)blockIdx.y * 128;
    const long nBlock = (long)blockIdx.x * 128;

    const unsigned long long aBase =
        (unsigned long long)(unsigned long)A + (unsigned long long)(mBlock * IN_F) * 2ull;
    const unsigned long long bBase =
        (unsigned long long)(unsigned long)Bw + (unsigned long long)(nBlock * IN_F) * 2ull;
    const unsigned ldsA0 = (unsigned)(unsigned long)&lsA[0][0];
    const unsigned ldsA1 = (unsigned)(unsigned long)&lsA[1][0];
    const unsigned ldsB0 = (unsigned)(unsigned long)&lsB[0][0];
    const unsigned ldsB1 = (unsigned)(unsigned long)&lsB[1][0];

    v8f acc[4][2];
    #pragma unroll
    for (int i = 0; i < 4; ++i)
        #pragma unroll
        for (int j = 0; j < 2; ++j)
            #pragma unroll
            for (int e = 0; e < 8; ++e) acc[i][j][e] = 0.0f;

    // Prologue: stream tile 0 into buffer 0 (wave 0 drives the TDM).
    if (wave == 0) {
        tdm_load_tile(aBase, ldsA0, (unsigned)MROWS);
        tdm_load_tile(bBase, ldsB0, (unsigned)NF);
    }

    const int NIT = IN_F / KT;   // 32
    for (int it = 0; it < NIT; ++it) {
        const int cur = it & 1;

        // All readers of the *other* buffer (iteration it-1) are done.
        __syncthreads();

        if (wave == 0) {
            if (it + 1 < NIT) {
                const unsigned long long koff = (unsigned long long)(it + 1) * KT * 2ull;
                tdm_load_tile(aBase + koff, cur ? ldsA0 : ldsA1, (unsigned)MROWS);
                tdm_load_tile(bBase + koff, cur ? ldsB0 : ldsB1, (unsigned)NF);
                // In-order TENSORcnt: <=2 outstanding => current tile landed.
                __builtin_amdgcn_s_wait_tensorcnt(2);
            } else {
                __builtin_amdgcn_s_wait_tensorcnt(0);
            }
        }
        __syncthreads();

        const bf16_t* bufA = lsA[cur];
        const bf16_t* bufB = lsB[cur];

        #pragma unroll
        for (int kk = 0; kk < 2; ++kk) {
            // A 16x32 fragment: lanes 0-15 (M=l16): K 0..7 + 16..23; lanes 16-31: +8
            v16bf afrag[4];
            #pragma unroll
            for (int i = 0; i < 4; ++i) {
                const int row = waveM * 64 + i * 16 + l16;
                v8bf lo = *(const v8bf*)(bufA + row * LDSS + kk * 32 + half * 8);
                v8bf hi = *(const v8bf*)(bufA + row * LDSS + kk * 32 + 16 + half * 8);
                #pragma unroll
                for (int e = 0; e < 8; ++e) { afrag[i][e] = lo[e]; afrag[i][8 + e] = hi[e]; }
            }
            // B 32x16 fragment: column N=l16; lanes 0-15: K 0..7 + 16..23; lanes 16-31: +8
            v16bf bfrag[2];
            #pragma unroll
            for (int j = 0; j < 2; ++j) {
                const int row = waveN * 32 + j * 16 + l16;
                v8bf lo = *(const v8bf*)(bufB + row * LDSS + kk * 32 + half * 8);
                v8bf hi = *(const v8bf*)(bufB + row * LDSS + kk * 32 + 16 + half * 8);
                #pragma unroll
                for (int e = 0; e < 8; ++e) { bfrag[j][e] = lo[e]; bfrag[j][8 + e] = hi[e]; }
            }
            #pragma unroll
            for (int i = 0; i < 4; ++i)
                #pragma unroll
                for (int j = 0; j < 2; ++j)
                    acc[i][j] = __builtin_amdgcn_wmma_f32_16x16x32_bf16(
                        false, afrag[i], false, bfrag[j],
                        (short)0, acc[i][j], false, false);
        }
    }

    // Epilogue: bias add + store (C layout: m = v + 8*half, n = l16).
    #pragma unroll
    for (int j = 0; j < 2; ++j) {
        const long n = nBlock + waveN * 32 + j * 16 + l16;
        const float bsum = bias[n];
        #pragma unroll
        for (int i = 0; i < 4; ++i) {
            const long mBase = mBlock + waveM * 64 + i * 16 + half * 8;
            #pragma unroll
            for (int v = 0; v < 8; ++v)
                out[(mBase + v) * (long)NF + n] = acc[i][j][v] + bsum;
        }
    }
}

// ---------------------------------------------------------------------------
// Launch: quantize weights -> bf16, convert x -> bf16, TDM+WMMA GEMM.
// Workspace: [ wq_bf16 : NF*IN_F*2 B ][ x_bf16 : MROWS*IN_F*2 B ]  (~48 MB)
// ---------------------------------------------------------------------------
extern "C" void kernel_launch(void* const* d_in, const int* in_sizes, int n_in,
                              void* d_out, int out_size, void* d_ws, size_t ws_size,
                              hipStream_t stream) {
    (void)in_sizes; (void)n_in; (void)out_size; (void)ws_size;

    const float* x       = (const float*)d_in[0];
    const float* weightT = (const float*)d_in[1];
    const float* bias    = (const float*)d_in[2];
    const float* value   = (const float*)d_in[3];
    const float* minsc   = (const float*)d_in[4];
    const float* maxsc   = (const float*)d_in[5];
    float* out = (float*)d_out;

    bf16_t* wq_bf = (bf16_t*)d_ws;
    bf16_t* x_bf  = (bf16_t*)((char*)d_ws + (size_t)NF * IN_F * sizeof(bf16_t));

    quant_w_kernel<<<NGRP / 8, 256, 0, stream>>>(weightT, value, minsc, maxsc, wq_bf);
    cvt_x_kernel<<<(MROWS * IN_F) / (256 * 8), 256, 0, stream>>>(x, x_bf);

    dim3 grid(NF / 128, MROWS / 128);
    gemm_bf16_kernel<<<grid, 256, 0, stream>>>(x_bf, wq_bf, bias, out);
}
